// CombSub_46720654246510
// MI455X (gfx1250) — matrix operand order repaired and checked by reading, not compiled, not run on hardware
//
#include <hip/hip_runtime.h>
#include <hip/hip_bf16.h>
#include <math.h>

typedef _Float16 f16;
typedef __attribute__((ext_vector_type(16))) _Float16 v16h;
typedef __attribute__((ext_vector_type(8)))  float    v8f;

// ---------------- problem constants ----------------
constexpr int NB = 16;          // batch
constexpr int NF = 512;         // frames (== seq len L)
constexpr int ND = 256;         // model dim
constexpr int NT = 262144;      // samples per batch row (512*512)
constexpr float SRATE = 44100.0f;
constexpr int NROWS = NB * NF;  // 8192 GEMM rows

// ---------------- ws arena (float offsets) ----------------
constexpr size_t OFF_PREFIX = 0;                    // 8192
constexpr size_t OFF_COMB   = 8192;                 // 4194304
constexpr size_t OFF_BUF1   = 4202496;              // 2097152
constexpr size_t OFF_XA     = 6299648;              // 2097152
constexpr size_t OFF_LNB    = 8396800;              // 2097152
constexpr size_t OFF_QKV    = 10493952;             // 6291456
constexpr size_t OFF_ATTN   = 16785408;             // 2097152
constexpr size_t OFF_GLU1   = 18882560;             // 8388608
constexpr size_t OFF_GLU2   = 27271168;             // 4194304
constexpr size_t OFF_GLU3   = 31465472;             // 4194304
constexpr size_t OFF_IRH    = 35659776;             // 8372224 (8192*1022)
constexpr size_t OFF_PACK   = 44032000;             // 1310720 floats (2621440 halfs)
// aliases (lifetimes do not overlap)
constexpr size_t OFF_CTRLS  = OFF_QKV;              // 8388608 (QKV+ATTN)
constexpr size_t OFF_MAGC   = OFF_GLU1;             // 2097152
constexpr size_t OFF_MAGS   = OFF_GLU1 + 2097152;   // 2097152
constexpr size_t OFF_SRC    = OFF_GLU1 + 4194304;   // 4194304
constexpr size_t OFF_NP     = OFF_XA;               // 2097152
constexpr size_t OFF_IRAP   = OFF_GLU2;             // 4177920
constexpr size_t OFF_IRN    = OFF_GLU3;             // 4177920
constexpr size_t OFF_H1     = OFF_BUF1;             // 4194304 (BUF1+XA)
constexpr size_t OFF_KP     = OFF_GLU2;             // Kp: 2097152 halfs (1048576 f)
constexpr size_t OFF_VP     = OFF_GLU3;             // Vp: 2097152 halfs

// packed-weight layout (half offsets inside pack region)
constexpr size_t PK_CONV1 = 0;            // 3 * 65536
constexpr size_t PK_CONV2 = 196608;       // 3 * 65536
constexpr size_t PK_LAYER = 393216;       // per layer 655360: wqkv(196608), wo(65536), pw1(262144), pw2(131072)
constexpr size_t PK_OUTW  = 393216 + 3 * 655360;  // 262144

// ---------------- d_out layout (floats) ----------------
constexpr size_t OUT_SIG   = 0;
constexpr size_t OUT_PHASE = (size_t)NB * NT;            // 4194304
constexpr size_t OUT_HARM  = OUT_PHASE + (size_t)NB * NF;// 4202496
constexpr size_t OUT_NOISE = OUT_HARM + (size_t)NB * NT; // 8396800

// =====================================================================
// 1) per-frame phase increments + exclusive prefix scan (per batch row)
// =====================================================================
__global__ __launch_bounds__(512)
void frame_scan_kernel(const float* __restrict__ f0, float* __restrict__ prefix)
{
    int b = blockIdx.x;
    int i = threadIdx.x;
    __shared__ float sd[512];
    float f0i = f0[b * NF + i];
    float f0n = (i < NF - 1) ? f0[b * NF + i + 1] : f0i;
    float d = (512.0f * f0i + (f0n - f0i) * (511.0f * 0.5f)) / SRATE;
    sd[i] = d;
    __syncthreads();
    for (int st = 1; st < 512; st <<= 1) {
        float t = (i >= st) ? sd[i - st] : 0.0f;
        __syncthreads();
        sd[i] += t;
        __syncthreads();
    }
    prefix[b * NF + i] = sd[i] - d;   // exclusive prefix
}

// =====================================================================
// 2) wrapped phase, combtooth (sinc), phase_frames
// =====================================================================
__global__ __launch_bounds__(256)
void phase_comb_kernel(const float* __restrict__ f0, const float* __restrict__ prefix,
                       float* __restrict__ comb, float* __restrict__ phase_out)
{
    int b = blockIdx.y;
    int n = blockIdx.x * 256 + threadIdx.x;
    int i = n >> 9, k = n & 511;
    float f0i = f0[b * NF + i];
    float f0n = (i < NF - 1) ? f0[b * NF + i + 1] : f0i;
    float df = f0n - f0i;
    float part = ((float)(k + 1) * f0i + df * ((float)k * (float)(k + 1)) * (0.5f / 512.0f)) / SRATE;
    float xv = prefix[b * NF + i] + part;
    float xw = xv - rintf(xv);
    float f0s = f0i + df * (float)k * (1.0f / 512.0f);
    float y = SRATE * xw / (f0s + 1e-3f);
    float piy = 3.14159265358979f * y;
    float cb = (fabsf(y) < 1e-6f) ? 1.0f : (__sinf(piy) / piy);
    comb[(size_t)b * NT + n] = cb;
    if (k == 0) phase_out[b * NF + i] = 6.28318530717959f * xw;
}

// =====================================================================
// helper: build v16h from 4 float4s (elements 0..15 = r0..r3 in order)
// =====================================================================
__device__ inline v16h pack16(const float4 r0, const float4 r1,
                              const float4 r2, const float4 r3)
{
    v16h a;
    a[0]=(f16)r0.x; a[1]=(f16)r0.y; a[2]=(f16)r0.z; a[3]=(f16)r0.w;
    a[4]=(f16)r1.x; a[5]=(f16)r1.y; a[6]=(f16)r1.z; a[7]=(f16)r1.w;
    a[8]=(f16)r2.x; a[9]=(f16)r2.y; a[10]=(f16)r2.z; a[11]=(f16)r2.w;
    a[12]=(f16)r3.x; a[13]=(f16)r3.y; a[14]=(f16)r3.z; a[15]=(f16)r3.w;
    return a;
}

// =====================================================================
// weight pre-pack: f32 row-major (K x N) -> f16 WMMA-fragment order
//   Wp[((kt*ntiles + j)*32 + lane)*16 + h] = W[(kt*32 + kk)*N + j*16 + col]
//   col = lane&15 ; kk = (lane&16 ? 8 : 0) + (h&8 ? 8 : 0) + h
// =====================================================================
__global__ __launch_bounds__(256)
void pack_weight_kernel(const float* __restrict__ W, f16* __restrict__ Wp, int N, int K)
{
    int d = blockIdx.x * 256 + threadIdx.x;
    if (d >= N * K) return;
    int h = d & 15, lane = (d >> 4) & 31;
    int rest = d >> 9;
    int ntiles = N >> 4;
    int j = rest % ntiles, kt = rest / ntiles;
    int col = j * 16 + (lane & 15);
    int kk = ((lane & 16) ? 8 : 0) + ((h & 8) ? 8 : 0) + h;
    Wp[d] = (f16)W[(size_t)(kt * 32 + kk) * N + col];
}

// =====================================================================
// 3) LDS-free f16-WMMA GEMM on packed weights:
//    Y = [accY?Y:0] + shift(X)@W (+bias) (+resid)
//    64x64 block, 4 waves; A direct f32 global, B contiguous packed f16.
// =====================================================================
__global__ __launch_bounds__(128)
void gemm_wmma(const float* __restrict__ X, const f16* __restrict__ Wp,
               const float* bias, const float* resid, float* Y,
               int M, int N, int K, int rowShift, int seqLen, int accY)
{
    int tid = threadIdx.x;
    int wave = tid >> 5, lane = tid & 31;
    int bm = blockIdx.y * 64, bn = blockIdx.x * 64;
    int ntiles = N >> 4;

    v8f acc[4];
    int c0 = lane & 15;
    int rb = (lane < 16) ? 0 : 8;
    if (accY) {
        #pragma unroll
        for (int j = 0; j < 4; ++j)
            #pragma unroll
            for (int r = 0; r < 8; ++r)
                acc[j][r] = Y[(size_t)(bm + wave * 16 + rb + r) * N + bn + j * 16 + c0];
    } else {
        v8f z = {0.f,0.f,0.f,0.f,0.f,0.f,0.f,0.f};
        #pragma unroll
        for (int j = 0; j < 4; ++j) acc[j] = z;
    }

    // this lane's fixed A row (ISA 16-bit A layout: lane<16 -> kb=0, else kb=8)
    int grow = bm + wave * 16 + (lane & 15);
    int lr = (grow % seqLen) + rowShift;
    bool okA = (lr >= 0 && lr < seqLen);
    const float* xrow = X + (size_t)(grow + rowShift) * K;
    int akb = (lane < 16) ? 0 : 8;

    const f16* wfrag = Wp + ((size_t)(bn >> 4) * 32 + lane) * 16;  // kt=0, j=0 fragment
    size_t ktStride = (size_t)ntiles * 512;                        // halfs per k-tile

    const float4 z4 = make_float4(0.f, 0.f, 0.f, 0.f);
    for (int k0 = 0; k0 < K; k0 += 32) {
        const float* xp = xrow + k0 + akb;
        float4 r0 = okA ? *(const float4*)(xp + 0)  : z4;
        float4 r1 = okA ? *(const float4*)(xp + 4)  : z4;
        float4 r2 = okA ? *(const float4*)(xp + 16) : z4;
        float4 r3 = okA ? *(const float4*)(xp + 20) : z4;
        v16h a = pack16(r0, r1, r2, r3);
        #pragma unroll
        for (int j = 0; j < 4; ++j) {
            v16h bf = *(const v16h*)(wfrag + (size_t)j * 512);
            acc[j] = __builtin_amdgcn_wmma_f32_16x16x32_f16(
                false, a, false, bf, (short)0, acc[j], false, false);
        }
        wfrag += ktStride;
    }
    #pragma unroll
    for (int j = 0; j < 4; ++j) {
        int gc = bn + j * 16 + c0;
        #pragma unroll
        for (int r = 0; r < 8; ++r) {
            int gr = bm + wave * 16 + rb + r;
            float v = acc[j][r];
            if (bias)  v += bias[gc];
            if (resid) v += resid[(size_t)gr * N + gc];
            Y[(size_t)gr * N + gc] = v;
        }
    }
}

// =====================================================================
// 4) GroupNorm(4 groups) + LeakyReLU(0.01), in-place on (8192,256)
// =====================================================================
__global__ __launch_bounds__(256)
void groupnorm_leaky(float* x, const float* __restrict__ g, const float* __restrict__ bt)
{
    int b = blockIdx.x, grp = blockIdx.y;
    __shared__ float rs[256], rss[256];
    float s = 0.f, ss = 0.f;
    for (int e = threadIdx.x; e < NF * 64; e += 256) {
        int l = e >> 6, c = e & 63;
        float v = x[(size_t)(b * NF + l) * ND + grp * 64 + c];
        s += v; ss += v * v;
    }
    rs[threadIdx.x] = s; rss[threadIdx.x] = ss;
    __syncthreads();
    for (int st = 128; st > 0; st >>= 1) {
        if (threadIdx.x < st) { rs[threadIdx.x] += rs[threadIdx.x + st]; rss[threadIdx.x] += rss[threadIdx.x + st]; }
        __syncthreads();
    }
    float mean = rs[0] * (1.0f / 32768.0f);
    float var = rss[0] * (1.0f / 32768.0f) - mean * mean;
    float inv = rsqrtf(var + 1e-5f);
    for (int e = threadIdx.x; e < NF * 64; e += 256) {
        int l = e >> 6, c = e & 63;
        int ch = grp * 64 + c;
        size_t idx = (size_t)(b * NF + l) * ND + ch;
        float v = (x[idx] - mean) * inv * g[ch] + bt[ch];
        x[idx] = (v >= 0.f) ? v : 0.01f * v;
    }
}

// =====================================================================
// 5) LayerNorm rows of 256 (one wave per row)
// =====================================================================
__global__ __launch_bounds__(256)
void layernorm_kernel(const float* __restrict__ x, float* __restrict__ y,
                      const float* __restrict__ g, const float* __restrict__ bt)
{
    int row = blockIdx.x * 8 + (threadIdx.x >> 5);
    int lane = threadIdx.x & 31;
    const float* xr = x + (size_t)row * ND;
    float v[8]; float s = 0.f;
    #pragma unroll
    for (int j = 0; j < 8; ++j) { v[j] = xr[lane + 32 * j]; s += v[j]; }
    #pragma unroll
    for (int m = 16; m > 0; m >>= 1) s += __shfl_xor(s, m);
    float mean = s * (1.0f / 256.0f);
    float ss = 0.f;
    #pragma unroll
    for (int j = 0; j < 8; ++j) { float d = v[j] - mean; ss += d * d; }
    #pragma unroll
    for (int m = 16; m > 0; m >>= 1) ss += __shfl_xor(ss, m);
    float inv = rsqrtf(ss * (1.0f / 256.0f) + 1e-5f);
    #pragma unroll
    for (int j = 0; j < 8; ++j) {
        int c = lane + 32 * j;
        y[(size_t)row * ND + c] = (v[j] - mean) * inv * g[c] + bt[c];
    }
}

// =====================================================================
// 6) embeddings add
// =====================================================================
__global__ __launch_bounds__(256)
void embed_add_kernel(float* x, const float* __restrict__ f0, const float* __restrict__ ph,
                      const float* __restrict__ vol,
                      const float* f0w, const float* f0b, const float* phw,
                      const float* phb, const float* vw, const float* vb)
{
    int idx = blockIdx.x * 256 + threadIdx.x;
    int row = idx >> 8, c = idx & 255;
    float lf = log1pf(f0[row] * (1.0f / 700.0f));
    float pv = ph[row] * (1.0f / 3.14159265358979f);
    float vv = vol[row];
    x[idx] += lf * f0w[c] + f0b[c] + pv * phw[c] + phb[c] + vv * vw[c] + vb[c];
}

// =====================================================================
// K/V fragment pre-pack for attention (per layer):
//   Kp[(((b*8+hd)*32 + kt)*32 + lane)*16 + h]           = K[kt*16+c][bkb+h]
//   Vp[(((b*8+hd)*32 + kt*2+t2)*32 + lane)*16 + h]      = V[kt*32+bkb+h][t2*16+c]
// =====================================================================
__global__ __launch_bounds__(256)
void qkv_pack_kernel(const float* __restrict__ qkv, f16* __restrict__ Kp, f16* __restrict__ Vp)
{
    int d = blockIdx.x * 256 + threadIdx.x;     // 0 .. 2097151
    int h = d & 15, lane = (d >> 4) & 31;
    int c = lane & 15, bkb = ((lane & 16) ? 8 : 0) + ((h & 8) ? 8 : 0);
    {   // K
        int kt = (d >> 9) & 31, hd = (d >> 14) & 7, b = d >> 17;
        int key = kt * 16 + c;
        Kp[d] = (f16)qkv[(size_t)(b * 512 + key) * 768 + 256 + hd * 32 + bkb + h];
    }
    {   // V
        int t2 = (d >> 9) & 1, kt = (d >> 10) & 15, hd = (d >> 14) & 7, b = d >> 17;
        int vrow = kt * 32 + bkb + h;
        Vp[d] = (f16)qkv[(size_t)(b * 512 + vrow) * 768 + 512 + hd * 32 + t2 * 16 + c];
    }
}

// =====================================================================
// 7) attention: one wave per (batch, head, 16-query tile), WMMA scores+AV
// =====================================================================
__global__ __launch_bounds__(32)
void attn_kernel(const float* __restrict__ qkv, const f16* __restrict__ Kp,
                 const f16* __restrict__ Vp, float* __restrict__ out)
{
    __shared__ float Sc[16][512];
    __shared__ __align__(32) f16 Atf[16][32][16];   // [ktile][lane][h]
    int lane = threadIdx.x;
    int q0 = blockIdx.x * 16;
    int hd = blockIdx.y, b = blockIdx.z;
    const float* qp = qkv + (size_t)b * NF * 768 + hd * 32;

    // Q fragment (16x32): two aligned 8-float runs
    v16h a;
    {
        int r = lane & 15, kb = (lane < 16) ? 0 : 8;
        const float* qrow = qp + (size_t)(q0 + r) * 768 + kb;
        a = pack16(*(const float4*)(qrow + 0),  *(const float4*)(qrow + 4),
                   *(const float4*)(qrow + 16), *(const float4*)(qrow + 20));
    }
    const float scale = 0.17677669529663687f; // 1/sqrt(32)
    int c = lane & 15;
    int rb = (lane < 16) ? 0 : 8;
    const f16* kfrag = Kp + ((size_t)(b * 8 + hd) * 32 * 32 + lane) * 16;
    for (int kt = 0; kt < 32; ++kt) {
        v16h bf = *(const v16h*)(kfrag + (size_t)kt * 512);
        v8f s = {0.f,0.f,0.f,0.f,0.f,0.f,0.f,0.f};
        s = __builtin_amdgcn_wmma_f32_16x16x32_f16(false, a, false, bf, (short)0, s, false, false);
        #pragma unroll
        for (int r = 0; r < 8; ++r) Sc[rb + r][kt * 16 + c] = s[r] * scale;
    }
    __syncthreads();
    // softmax: lane l handles row l&15, half (l>>4); write fragment-ordered f16
    {
        int row = lane & 15, cb = (lane >> 4) * 256;
        float mx = -3.4e38f;
        for (int j = 0; j < 256; ++j) mx = fmaxf(mx, Sc[row][cb + j]);
        mx = fmaxf(mx, __shfl_xor(mx, 16));
        float sm = 0.f;
        for (int j = 0; j < 256; ++j) sm += __expf(Sc[row][cb + j] - mx);
        sm += __shfl_xor(sm, 16);
        float inv = 1.0f / sm;
        for (int j = 0; j < 256; ++j) {
            int k = cb + j;
            int kt = k >> 5, kl = k & 31;
            int hi = ((kl >= 8 && kl < 16) || kl >= 24) ? 1 : 0;
            int h = kl - ((kl >= 24) ? 16 : ((kl >= 8) ? 8 : 0));
            Atf[kt][row + 16 * hi][h] = (f16)(__expf(Sc[row][cb + j] - mx) * inv);
        }
    }
    __syncthreads();
    // O = att(16x512) @ V(512x32)
    v8f o0 = {0.f,0.f,0.f,0.f,0.f,0.f,0.f,0.f}, o1 = o0;
    const f16* vfrag = Vp + ((size_t)(b * 8 + hd) * 32 * 32 + lane) * 16;
    for (int kt = 0; kt < 16; ++kt) {
        v16h aa = *(const v16h*)(&Atf[kt][lane][0]);
        v16h b0 = *(const v16h*)(vfrag + (size_t)(kt * 2) * 512);
        v16h b1 = *(const v16h*)(vfrag + (size_t)(kt * 2 + 1) * 512);
        o0 = __builtin_amdgcn_wmma_f32_16x16x32_f16(false, aa, false, b0, (short)0, o0, false, false);
        o1 = __builtin_amdgcn_wmma_f32_16x16x32_f16(false, aa, false, b1, (short)0, o1, false, false);
    }
    float* op = out + ((size_t)b * NF + q0) * ND + hd * 32;
    #pragma unroll
    for (int r = 0; r < 8; ++r) {
        op[(size_t)(rb + r) * ND + c]      = o0[r];
        op[(size_t)(rb + r) * ND + 16 + c] = o1[r];
    }
}

// =====================================================================
// 8) GLU, depthwise-conv(k=31)+swish
// =====================================================================
__global__ __launch_bounds__(256)
void glu_kernel(const float* __restrict__ in, float* __restrict__ out)
{
    int idx = blockIdx.x * 256 + threadIdx.x;   // 8192*512
    int row = idx >> 9, cc = idx & 511;
    float av = in[(size_t)row * 1024 + cc];
    float gv = in[(size_t)row * 1024 + 512 + cc];
    out[idx] = av * (1.0f / (1.0f + __expf(-gv)));
}

__global__ __launch_bounds__(256)
void dwconv_swish_kernel(const float* __restrict__ in, float* __restrict__ out,
                         const float* __restrict__ w, const float* __restrict__ bias)
{
    int idx = blockIdx.x * 256 + threadIdx.x;   // 8192*512
    int row = idx >> 9, cc = idx & 511;
    int l = row & 511;
    int rowbase = row - l;
    float acc = bias[cc];
    #pragma unroll
    for (int t = 0; t < 31; ++t) {
        int ll = l + t - 15;
        if (ll >= 0 && ll < 512)
            acc += in[(size_t)(rowbase + ll) * 512 + cc] * w[t * 512 + cc];
    }
    out[idx] = acc * (1.0f / (1.0f + __expf(-acc)));
}

// =====================================================================
// 9) ctrl post-processing
// =====================================================================
__global__ __launch_bounds__(256)
void magap_kernel(const float* __restrict__ ctrls, float* __restrict__ cp, float* __restrict__ sp)
{
    int f = blockIdx.x * 256 + threadIdx.x;     // 8192 frames
    const float* c = ctrls + (size_t)f * 1024;
    float acc = 0.f;
    for (int k = 0; k < 256; ++k) {
        acc += 3.14159265358979f * tanhf(c[k]);
        float s, co; __sincosf(acc, &s, &co);
        cp[(size_t)f * 256 + k] = co;
        sp[(size_t)f * 256 + k] = s;
    }
}

__global__ __launch_bounds__(256)
void expparam_kernel(const float* __restrict__ ctrls, float* __restrict__ src, float* __restrict__ np_)
{
    int idx = blockIdx.x * 256 + threadIdx.x;   // 8192*768
    int f = idx / 768, cc = idx % 768;
    if (cc < 512) src[(size_t)f * 512 + cc] = __expf(ctrls[(size_t)f * 1024 + 256 + cc]);
    else          np_[(size_t)f * 256 + (cc - 512)] = __expf(ctrls[(size_t)f * 1024 + 768 + (cc - 512)]) * (1.0f / 128.0f);
}

// =====================================================================
// 10) IR build: direct DFT (irfft + roll(win/2) + optional window fused)
//     wmode: 0=none, 1=dynamic(f0), 2=hann
// =====================================================================
__global__ __launch_bounds__(256)
void ir_dft_kernel(const float* __restrict__ Xre, const float* Xim, float* __restrict__ ir,
                   int nmag, int win, int wmode, const float* f0f)
{
    int f = blockIdx.y;
    int t = blockIdx.x * 256 + threadIdx.x;
    if (t >= win) return;
    const float* xr = Xre + (size_t)f * nmag;
    const float* xi = Xim ? (Xim + (size_t)f * nmag) : nullptr;
    int half = win >> 1;
    float phi = 6.28318530717959f * (float)(t - half) / (float)win;
    float sn, cs; __sincosf(phi, &sn, &cs);
    float curr = cs, curi = sn;   // e^{i*phi*k}, k=1
    float accm = 0.f;
    for (int k = 1; k < nmag - 1; ++k) {
        accm += xr[k] * curr;
        if (xi) accm -= xi[k] * curi;
        float nr = curr * cs - curi * sn;
        curi = curr * sn + curi * cs;
        curr = nr;
    }
    float nyq = xr[nmag - 1] * (((t + half) & 1) ? -1.0f : 1.0f);
    float v = (xr[0] + nyq + 2.0f * accm) / (float)win;
    float w = 1.0f;
    if (wmode == 1) {
        float hw = 1.5f * SRATE / (f0f[f] + 1e-3f);
        float pos = (float)(t - half);
        w = (fabsf(pos) < hw) ? (0.5f + 0.5f * __cosf(3.14159265358979f * pos / hw)) : 0.0f;
    } else if (wmode == 2) {
        w = 0.5f - 0.5f * __cosf(6.28318530717959f * (float)t / (float)win);
    }
    ir[(size_t)f * win + t] = v * w;
}

// =====================================================================
// 11) LTV direct convolution (== fft_convolve + OLA + crop)
// =====================================================================
__global__ __launch_bounds__(256)
void ltv_conv_kernel(const float* __restrict__ audio, const float* __restrict__ ir,
                     float* __restrict__ out, int win)
{
    __shared__ float smem[256 + 1022];
    int b = blockIdx.y;
    int n0 = blockIdx.x * 256;
    int half = win >> 1;
    int lo = n0 - half + 1;
    for (int e = threadIdx.x; e < 255 + win; e += 256) {
        int m = lo + e;
        smem[e] = (m >= 0 && m < NT) ? audio[(size_t)b * NT + m] : 0.0f;
    }
    const float* irb = ir + (size_t)b * NF * win;
    {   // prefetch first IR row this block touches (global_prefetch_b8)
        int m0 = lo < 0 ? 0 : lo;
        __builtin_prefetch(irb + (size_t)(m0 >> 9) * win, 0, 1);
    }
    __syncthreads();
    int n = n0 + threadIdx.x;
    float acc = 0.f;
    for (int j = 0; j < win; ++j) {
        int m = n - half + 1 + j;
        if ((unsigned)m < (unsigned)NT) {
            int fr = m >> 9;
            acc += smem[threadIdx.x + j] * irb[(size_t)fr * win + (win - 1 - j)];
        }
    }
    out[(size_t)b * NT + n] = acc;
}

__global__ __launch_bounds__(256)
void add_kernel(const float* __restrict__ aa, const float* __restrict__ bb, float* __restrict__ cc)
{
    size_t i = (size_t)blockIdx.x * 256 + threadIdx.x;
    cc[i] = aa[i] + bb[i];
}

// =====================================================================
// host
// =====================================================================
static inline void launch_gemm(const float* X, const f16* Wp, const float* bias,
                               const float* resid, float* Y, int M, int N, int K,
                               int rowShift, int accY, hipStream_t s)
{
    dim3 g(N / 64, M / 64);
    gemm_wmma<<<g, 128, 0, s>>>(X, Wp, bias, resid, Y, M, N, K, rowShift, NF, accY);
}

static inline void launch_pack(const float* W, f16* Wp, int N, int K, hipStream_t s)
{
    int n = N * K;
    pack_weight_kernel<<<(n + 255) / 256, 256, 0, s>>>(W, Wp, N, K);
}

extern "C" void kernel_launch(void* const* d_in, const int* in_sizes, int n_in,
                              void* d_out, int out_size, void* d_ws, size_t ws_size,
                              hipStream_t stream)
{
    (void)in_sizes; (void)n_in; (void)out_size; (void)ws_size;
    const float* units = (const float*)d_in[0];
    const float* f0f   = (const float*)d_in[1];
    const float* volf  = (const float*)d_in[2];
    const float* noise = (const float*)d_in[3];
    // params, insertion-order flatten
    const float* conv1_w = (const float*)d_in[4];
    const float* conv1_b = (const float*)d_in[5];
    const float* gn_g    = (const float*)d_in[6];
    const float* gn_b    = (const float*)d_in[7];
    const float* conv2_w = (const float*)d_in[8];
    const float* conv2_b = (const float*)d_in[9];
    const float* f0_w = (const float*)d_in[10];
    const float* f0_b = (const float*)d_in[11];
    const float* ph_w = (const float*)d_in[12];
    const float* ph_b = (const float*)d_in[13];
    const float* vol_w = (const float*)d_in[14];
    const float* vol_b = (const float*)d_in[15];
    const int LW0 = 16, LSTRIDE = 14;
    const float* out_ln_g = (const float*)d_in[LW0 + 3 * LSTRIDE + 0];
    const float* out_ln_b = (const float*)d_in[LW0 + 3 * LSTRIDE + 1];
    const float* out_w    = (const float*)d_in[LW0 + 3 * LSTRIDE + 2];
    const float* out_b    = (const float*)d_in[LW0 + 3 * LSTRIDE + 3];

    float* ws = (float*)d_ws;
    float* outp = (float*)d_out;
    float* prefix = ws + OFF_PREFIX;
    float* comb   = ws + OFF_COMB;
    float* buf1   = ws + OFF_BUF1;
    float* xa     = ws + OFF_XA;
    float* lnb    = ws + OFF_LNB;
    float* qkv    = ws + OFF_QKV;
    float* attn   = ws + OFF_ATTN;
    float* glu1   = ws + OFF_GLU1;
    float* glu2   = ws + OFF_GLU2;
    float* glu3   = ws + OFF_GLU3;
    float* ctrls  = ws + OFF_CTRLS;
    float* magc   = ws + OFF_MAGC;
    float* mags   = ws + OFF_MAGS;
    float* srcp   = ws + OFF_SRC;
    float* noisep = ws + OFF_NP;
    float* ir_ap  = ws + OFF_IRAP;
    float* ir_h   = ws + OFF_IRH;
    float* ir_n   = ws + OFF_IRN;
    float* harm1  = ws + OFF_H1;
    f16*   packb  = (f16*)(ws + OFF_PACK);
    f16*   Kp     = (f16*)(ws + OFF_KP);
    f16*   Vp     = (f16*)(ws + OFF_VP);

    // --- weight pre-pack (fragment-ordered f16) ---
    for (int t = 0; t < 3; ++t) {
        launch_pack(conv1_w + (size_t)t * ND * ND, packb + PK_CONV1 + (size_t)t * 65536, ND, ND, stream);
        launch_pack(conv2_w + (size_t)t * ND * ND, packb + PK_CONV2 + (size_t)t * 65536, ND, ND, stream);
    }
    for (int lyr = 0; lyr < 3; ++lyr) {
        const int p = LW0 + lyr * LSTRIDE;
        f16* base = packb + PK_LAYER + (size_t)lyr * 655360;
        launch_pack((const float*)d_in[p + 2],  base,          768,  ND, stream);   // wqkv
        launch_pack((const float*)d_in[p + 4],  base + 196608, ND,   ND, stream);   // wo
        launch_pack((const float*)d_in[p + 8],  base + 262144, 1024, ND, stream);   // pw1
        launch_pack((const float*)d_in[p + 12], base + 524288, ND,  512, stream);   // pw2
    }
    launch_pack(out_w, packb + PK_OUTW, 1024, ND, stream);

    // --- phase / combtooth ---
    frame_scan_kernel<<<NB, 512, 0, stream>>>(f0f, prefix);
    phase_comb_kernel<<<dim3(NT / 256, NB), 256, 0, stream>>>(f0f, prefix, comb, outp + OUT_PHASE);

    // --- unit2ctrl ---
    for (int t = 0; t < 3; ++t)
        launch_gemm(units, packb + PK_CONV1 + (size_t)t * 65536, (t == 0) ? conv1_b : nullptr,
                    nullptr, buf1, NROWS, ND, ND, t - 1, t != 0, stream);
    groupnorm_leaky<<<dim3(NB, 4), 256, 0, stream>>>(buf1, gn_g, gn_b);
    for (int t = 0; t < 3; ++t)
        launch_gemm(buf1, packb + PK_CONV2 + (size_t)t * 65536, (t == 0) ? conv2_b : nullptr,
                    nullptr, xa, NROWS, ND, ND, t - 1, t != 0, stream);
    embed_add_kernel<<<NROWS, 256, 0, stream>>>(xa, f0f, outp + OUT_PHASE, volf,
                                                f0_w, f0_b, ph_w, ph_b, vol_w, vol_b);
    for (int lyr = 0; lyr < 3; ++lyr) {
        const int p = LW0 + lyr * LSTRIDE;
        const float* ln_g   = (const float*)d_in[p + 0];
        const float* ln_b   = (const float*)d_in[p + 1];
        const float* bqkv   = (const float*)d_in[p + 3];
        const float* bo     = (const float*)d_in[p + 5];
        const float* ccln_g = (const float*)d_in[p + 6];
        const float* ccln_b = (const float*)d_in[p + 7];
        const float* pw1_b  = (const float*)d_in[p + 9];
        const float* dw_w   = (const float*)d_in[p + 10];
        const float* dw_b   = (const float*)d_in[p + 11];
        const float* pw2_b  = (const float*)d_in[p + 13];
        f16* base = packb + PK_LAYER + (size_t)lyr * 655360;

        layernorm_kernel<<<NROWS / 8, 256, 0, stream>>>(xa, lnb, ln_g, ln_b);
        launch_gemm(lnb, base, bqkv, nullptr, qkv, NROWS, 768, ND, 0, 0, stream);
        qkv_pack_kernel<<<2097152 / 256, 256, 0, stream>>>(qkv, Kp, Vp);
        attn_kernel<<<dim3(NF / 16, 8, NB), 32, 0, stream>>>(qkv, Kp, Vp, attn);
        launch_gemm(attn, base + 196608, bo, xa, xa, NROWS, ND, ND, 0, 0, stream);

        layernorm_kernel<<<NROWS / 8, 256, 0, stream>>>(xa, lnb, ccln_g, ccln_b);
        launch_gemm(lnb, base + 262144, pw1_b, nullptr, glu1, NROWS, 1024, ND, 0, 0, stream);
        glu_kernel<<<NROWS * 512 / 256, 256, 0, stream>>>(glu1, glu2);
        dwconv_swish_kernel<<<NROWS * 512 / 256, 256, 0, stream>>>(glu2, glu3, dw_w, dw_b);
        launch_gemm(glu3, base + 524288, pw2_b, xa, xa, NROWS, ND, 512, 0, 0, stream);
    }
    layernorm_kernel<<<NROWS / 8, 256, 0, stream>>>(xa, lnb, out_ln_g, out_ln_b);
    launch_gemm(lnb, packb + PK_OUTW, out_b, nullptr, ctrls, NROWS, 1024, ND, 0, 0, stream);

    // --- control -> filter magnitudes / IRs ---
    magap_kernel<<<NROWS / 256, 256, 0, stream>>>(ctrls, magc, mags);
    expparam_kernel<<<NROWS * 768 / 256, 256, 0, stream>>>(ctrls, srcp, noisep);
    ir_dft_kernel<<<dim3(2, NROWS), 256, 0, stream>>>(magc, mags, ir_ap, 256, 510, 0, f0f);
    ir_dft_kernel<<<dim3(4, NROWS), 256, 0, stream>>>(srcp, nullptr, ir_h, 512, 1022, 1, f0f);
    ir_dft_kernel<<<dim3(2, NROWS), 256, 0, stream>>>(noisep, nullptr, ir_n, 256, 510, 2, f0f);

    // --- LTV filtering chain ---
    ltv_conv_kernel<<<dim3(NT / 256, NB), 256, 0, stream>>>(comb, ir_ap, harm1, 510);
    ltv_conv_kernel<<<dim3(NT / 256, NB), 256, 0, stream>>>(harm1, ir_h, outp + OUT_HARM, 1022);
    ltv_conv_kernel<<<dim3(NT / 256, NB), 256, 0, stream>>>(noise, ir_n, outp + OUT_NOISE, 510);
    add_kernel<<<(NB * NT) / 256, 256, 0, stream>>>(outp + OUT_HARM, outp + OUT_NOISE, outp + OUT_SIG);
}
